// ModelClass_58643483460156
// MI455X (gfx1250) — matrix operand order repaired and checked by reading, not compiled, not run on hardware
//
#include <hip/hip_runtime.h>
#include <hip/hip_bf16.h>

typedef __attribute__((ext_vector_type(16))) _Float16 v16h;
typedef __attribute__((ext_vector_type(8)))  _Float16 v8h;
typedef __attribute__((ext_vector_type(8)))  float    v8f;

#define LEAKY_NEG 0.2f

__device__ inline float leaky(float v) {
  return fmaxf(v, 0.f) + LEAKY_NEG * fminf(v, 0.f);
}

// ---------------------------------------------------------------------------
// WMMA helpers (CDNA5: wave32, v_wmma_f32_16x16x32_f16)
// A: 16x32 f16, lane m/m+16 holds row m, two contiguous 8-half runs.
// B: 32x16 f16, pre-shuffled "fragment layout": lane-contiguous 16 halves.
// C: 16x16 f32, lane L holds col (L&15), rows r + 8*(L>=16), r=0..7.
// ---------------------------------------------------------------------------

// Stage fp32 weights W[K][N] into LDS as WMMA-B fragments:
// frag(nt, kc) is 512 halves; element (lane, j) = W[kc*32 + 16*(lane>=16)+j][nt*16 + (lane&15)]
__device__ inline void stage_w_frag(const float* W, int K, int N,
                                    _Float16* Wl, int KP, int NP,
                                    int tid, int nth) {
  const int kchunks = KP >> 5;
  const int total = (NP >> 4) * kchunks * 512;
  for (int i = tid; i < total; i += nth) {
    int frag = i >> 9;
    int lane = (i >> 4) & 31;
    int j = i & 15;
    int nt = frag / kchunks;
    int kc = frag - nt * kchunks;
    int k = (kc << 5) + ((lane >> 4) << 4) + j;
    int n = (nt << 4) + (lane & 15);
    Wl[i] = (k < K && n < N) ? (_Float16)W[k * N + n] : (_Float16)0.f;
  }
}
__device__ inline void stage_b(const float* b, int N, float* bl, int NP,
                               int tid, int nth) {
  for (int i = tid; i < NP; i += nth) bl[i] = (i < N) ? b[i] : 0.f;
}

// One FFN layer on a 16-row tile: out[16][NP] = act(in[16][KP] @ W[KP][NP] + b)
__device__ inline void ffn_layer16(const _Float16* inT, int KP,
                                   const _Float16* Wfrag,   // fragment layout
                                   const float* bl,
                                   int NP, int Nreal, bool do_leaky,
                                   _Float16* outT,          // f16 out tile [16][NP] (or null)
                                   float* gOut, int gLd,    // optional f32 rows out (or null)
                                   int lane) {
  const int nloc  = lane & 15;
  const int half  = lane >> 4;
  const int rbase = half << 3;
  const int m     = lane & 15;
  const int kchunks = KP >> 5;
  for (int nt = 0; nt < (NP >> 4); ++nt) {
    v8f c = {};
    for (int kci = 0; kci < kchunks; ++kci) {
      const int kc = kci << 5;
      v8h lo = *(const v8h*)(inT + m * KP + kc + (half << 4));
      v8h hi = *(const v8h*)(inT + m * KP + kc + 8 + (half << 4));
      v16h a, b;
#pragma unroll
      for (int j = 0; j < 8; ++j) { a[j] = lo[j]; a[8 + j] = hi[j]; }
      b = *(const v16h*)(Wfrag + ((nt * kchunks + kci) << 9) + (lane << 4));
      c = __builtin_amdgcn_wmma_f32_16x16x32_f16(false, a, false, b,
                                                 (short)0, c, false, false);
    }
    int col = (nt << 4) + nloc;
    float bias = bl[col];
#pragma unroll
    for (int r = 0; r < 8; ++r) {
      float v = c[r] + bias;
      if (do_leaky) v = leaky(v);
      int row = rbase + r;
      if (outT) outT[row * NP + col] = (_Float16)v;
      if (gOut && col < Nreal) gOut[row * gLd + col] = v;
    }
  }
}

// ---------------------------------------------------------------------------
// Kernel 1: per-point conv MLP (3->16->32->64, leaky) + max-pool over 2048 pts
// ---------------------------------------------------------------------------
__global__ __launch_bounds__(256) void k_conv_max(
    const float* __restrict__ x,
    const float* __restrict__ W0, const float* __restrict__ b0,
    const float* __restrict__ W1, const float* __restrict__ b1,
    const float* __restrict__ W2, const float* __restrict__ b2,
    float* __restrict__ feat) {
  __shared__ float sW0[48], sB0[16], sW1[512], sB1[32], sW2[2048], sB2[64];
  __shared__ float sred[8][64];
  int tid = threadIdx.x, b = blockIdx.x;
  for (int i = tid; i < 48; i += 256) sW0[i] = W0[i];
  for (int i = tid; i < 16; i += 256) sB0[i] = b0[i];
  for (int i = tid; i < 512; i += 256) sW1[i] = W1[i];
  for (int i = tid; i < 32; i += 256) sB1[i] = b1[i];
  for (int i = tid; i < 2048; i += 256) sW2[i] = W2[i];
  for (int i = tid; i < 64; i += 256) sB2[i] = b2[i];
  __syncthreads();
  float fm[64];
  for (int d = 0; d < 64; ++d) fm[d] = -3.402823466e38f;
  for (int n = tid; n < 2048; n += 256) {
    long long g = (long long)b * 2048 + n;
    float x0 = x[g * 3], x1 = x[g * 3 + 1], x2 = x[g * 3 + 2];
    float h1[16];
    for (int o = 0; o < 16; ++o)
      h1[o] = leaky(sB0[o] + x0 * sW0[o] + x1 * sW0[16 + o] + x2 * sW0[32 + o]);
    float h2[32];
    for (int o = 0; o < 32; ++o) {
      float v = sB1[o];
      for (int i = 0; i < 16; ++i) v += h1[i] * sW1[i * 32 + o];
      h2[o] = leaky(v);
    }
    for (int o = 0; o < 64; ++o) {
      float v = sB2[o];
      for (int i = 0; i < 32; ++i) v += h2[i] * sW2[i * 64 + o];
      fm[o] = fmaxf(fm[o], leaky(v));
    }
  }
  for (int d = 0; d < 64; ++d)
    for (int off = 16; off > 0; off >>= 1)
      fm[d] = fmaxf(fm[d], __shfl_xor(fm[d], off, 32));
  int w = tid >> 5, lane = tid & 31;
  if (lane == 0)
    for (int d = 0; d < 64; ++d) sred[w][d] = fm[d];
  __syncthreads();
  if (tid < 64) {
    float m = sred[0][tid];
    for (int ww = 1; ww < 8; ++ww) m = fmaxf(m, sred[ww][tid]);
    feat[b * 64 + tid] = m;
  }
}

// ---------------------------------------------------------------------------
// Kernel 2: rgan FFN (64 -> 64 leaky -> 4 linear), 32 rows
// ---------------------------------------------------------------------------
__global__ __launch_bounds__(64) void k_rgan(
    const float* __restrict__ feat,
    const float* __restrict__ W0, const float* __restrict__ b0,
    const float* __restrict__ W1, const float* __restrict__ b1,
    float* __restrict__ rgan) {
  __shared__ float fin[64], hid[64];
  int b = blockIdx.x, t = threadIdx.x;
  fin[t] = feat[b * 64 + t];
  __syncthreads();
  float v = b0[t];
  for (int i = 0; i < 64; ++i) v += fin[i] * W0[i * 64 + t];
  hid[t] = leaky(v);
  __syncthreads();
  if (t < 4) {
    float o = b1[t];
    for (int i = 0; i < 64; ++i) o += hid[i] * W1[i * 4 + t];
    rgan[b * 4 + t] = o;
  }
}

// ---------------------------------------------------------------------------
// Kernel 3: lin FFN per node (8 -> 64 -> 64 -> 30), WMMA
// ---------------------------------------------------------------------------
__global__ __launch_bounds__(256) void k_lin(
    const float* __restrict__ x, const float* __restrict__ cond,
    const float* __restrict__ rgan,
    const float* W0, const float* b0, const float* W1, const float* b1,
    const float* W2, const float* b2, float* __restrict__ hA) {
  __shared__ __attribute__((aligned(32))) _Float16 sW0[32 * 64], sW1[64 * 64], sW2[64 * 32];
  __shared__ float sB0[64], sB1[64], sB2[32];
  __shared__ __attribute__((aligned(32))) _Float16 tA[8][16 * 64], tB[8][16 * 64];
  int tid = threadIdx.x, lane = tid & 31, w = tid >> 5;
  stage_w_frag(W0, 8, 64, sW0, 32, 64, tid, 256);
  stage_w_frag(W1, 64, 64, sW1, 64, 64, tid, 256);
  stage_w_frag(W2, 64, 30, sW2, 64, 32, tid, 256);
  stage_b(b0, 64, sB0, 64, tid, 256);
  stage_b(b1, 64, sB1, 64, tid, 256);
  stage_b(b2, 30, sB2, 32, tid, 256);
  __syncthreads();
  int g0 = blockIdx.x * 128 + w * 16;
  if (lane < 16) {
    int g = g0 + lane;
    int b = g >> 11;
    _Float16* row = &tA[w][lane * 32];
    row[0] = (_Float16)x[(long long)g * 3 + 0];
    row[1] = (_Float16)x[(long long)g * 3 + 1];
    row[2] = (_Float16)x[(long long)g * 3 + 2];
    row[3] = (_Float16)cond[b];
    row[4] = (_Float16)rgan[b * 4 + 0];
    row[5] = (_Float16)rgan[b * 4 + 1];
    row[6] = (_Float16)rgan[b * 4 + 2];
    row[7] = (_Float16)rgan[b * 4 + 3];
    for (int k = 8; k < 32; ++k) row[k] = (_Float16)0.f;
  }
  __syncthreads();
  ffn_layer16(&tA[w][0], 32, sW0, sB0, 64, 64, true, &tB[w][0], nullptr, 0, lane);
  __syncthreads();
  ffn_layer16(&tB[w][0], 64, sW1, sB1, 64, 64, true, &tA[w][0], nullptr, 0, lane);
  __syncthreads();
  ffn_layer16(&tA[w][0], 64, sW2, sB2, 32, 30, false, nullptr,
              hA + (long long)g0 * 30, 30, lane);
}

// ---------------------------------------------------------------------------
// Kernel 4: prep f16 copy of h (K padded to 32) + squared norms
// ---------------------------------------------------------------------------
__global__ __launch_bounds__(256) void k_prep(const float* __restrict__ hA,
                                              _Float16* __restrict__ h16,
                                              float* __restrict__ sq) {
  long long g = (long long)blockIdx.x * 256 + threadIdx.x;
  float s = 0.f;
  for (int d = 0; d < 30; ++d) {
    float v = hA[g * 30 + d];
    s += v * v;
    h16[g * 32 + d] = (_Float16)v;
  }
  h16[g * 32 + 30] = (_Float16)0.f;
  h16[g * 32 + 31] = (_Float16)0.f;
  sq[g] = s;
}

// ---------------------------------------------------------------------------
// Kernel 5: kNN. Gram matrix G = h h^T via WMMA, rank by sq_j - 2G, top-8.
// Block = 16 query rows of one batch, 8 waves sweep columns in chunks of 512.
// ---------------------------------------------------------------------------
__global__ __launch_bounds__(256) void k_knn(const _Float16* __restrict__ h16,
                                             const float* __restrict__ sq,
                                             int* __restrict__ knn) {
  __shared__ float sc[16 * 512];
  __shared__ float cval[16 * 128];
  __shared__ int   cidx[16 * 128];
  __shared__ float bval[16 * 8];
  __shared__ int   bidx[16 * 8];
  int tid = threadIdx.x, lane = tid & 31, w = tid >> 5;
  int b = blockIdx.x >> 7, rt = blockIdx.x & 127;
  int i0 = rt * 16;
  long long base = (long long)b * 2048;
  int m = lane & 15, half = lane >> 4;
  // A fragment: rows i0..i0+15, k = 16*half + j (one contiguous 32B run per lane)
  v8h alo = *(const v8h*)(h16 + (base + i0 + m) * 32 + (half << 4));
  v8h ahi = *(const v8h*)(h16 + (base + i0 + m) * 32 + 8 + (half << 4));
  v16h a;
#pragma unroll
  for (int j = 0; j < 8; ++j) { a[j] = alo[j]; a[8 + j] = ahi[j]; }
  if (tid < 16)
    for (int s = 0; s < 8; ++s) { bval[tid * 8 + s] = 3.0e38f; bidx[tid * 8 + s] = 0; }
  __syncthreads();
  int n = lane & 15;
  for (int c = 0; c < 2048; c += 512) {
    for (int t = 0; t < 4; ++t) {               // 32 column tiles / 8 waves
      int j0 = c + (w * 4 + t) * 16;
      // B fragment: one aligned 32B contiguous load per lane
      v16h bf = *(const v16h*)(h16 + (base + j0 + n) * 32 + (half << 4));
      if (t < 3)  // prefetch next column tile (global_prefetch_b8)
        __builtin_prefetch(h16 + (base + j0 + 16 + n) * 32 + (half << 4), 0, 1);
      v8f acc = {};
      acc = __builtin_amdgcn_wmma_f32_16x16x32_f16(false, a, false, bf,
                                                   (short)0, acc, false, false);
      int col = j0 + n;
      float sqc = sq[base + col];
      int rbase = half << 3;
#pragma unroll
      for (int r = 0; r < 8; ++r)
        sc[(rbase + r) * 512 + (col - c)] = sqc - 2.0f * acc[r];
    }
    __syncthreads();
    {   // per-strip local top-8 (16 threads per row, 32 cols each)
      int row = tid >> 4, sub = tid & 15;
      float v[8]; int ix[8];
      for (int s = 0; s < 8; ++s) { v[s] = 3.0e38f; ix[s] = 0; }
      int cb = sub * 32;
      for (int e = 0; e < 32; ++e) {
        float sv = sc[row * 512 + cb + e];
        if (sv < v[7]) {
          int p = 7;
          while (p > 0 && v[p - 1] > sv) { v[p] = v[p - 1]; ix[p] = ix[p - 1]; --p; }
          v[p] = sv; ix[p] = c + cb + e;
        }
      }
      for (int s = 0; s < 8; ++s) {
        cval[row * 128 + sub * 8 + s] = v[s];
        cidx[row * 128 + sub * 8 + s] = ix[s];
      }
    }
    __syncthreads();
    if (tid < 16) {   // merge 128 candidates into running best-8
      int row = tid;
      float v[8]; int ix[8];
      for (int s = 0; s < 8; ++s) { v[s] = bval[row * 8 + s]; ix[s] = bidx[row * 8 + s]; }
      for (int e = 0; e < 128; ++e) {
        float sv = cval[row * 128 + e];
        if (sv < v[7]) {
          int si = cidx[row * 128 + e];
          int p = 7;
          while (p > 0 && v[p - 1] > sv) { v[p] = v[p - 1]; ix[p] = ix[p - 1]; --p; }
          v[p] = sv; ix[p] = si;
        }
      }
      for (int s = 0; s < 8; ++s) { bval[row * 8 + s] = v[s]; bidx[row * 8 + s] = ix[s]; }
    }
    __syncthreads();
  }
  if (tid < 16)
    for (int s = 0; s < 8; ++s)
      knn[(base + i0 + tid) * 8 + s] = bidx[tid * 8 + s];
}

// ---------------------------------------------------------------------------
// Kernel 6: GIN layer: in = h + sum_k h[nbr], FFN 30->64->64->30, WMMA
// ---------------------------------------------------------------------------
__global__ __launch_bounds__(256) void k_gin(
    const float* __restrict__ hIn, const int* __restrict__ knn,
    const float* W0, const float* b0, const float* W1, const float* b1,
    const float* W2, const float* b2, float* __restrict__ hOut) {
  __shared__ __attribute__((aligned(32))) _Float16 sW0[32 * 64], sW1[64 * 64], sW2[64 * 32];
  __shared__ float sB0[64], sB1[64], sB2[32];
  __shared__ __attribute__((aligned(32))) _Float16 tA[8][16 * 64], tB[8][16 * 64];
  int tid = threadIdx.x, lane = tid & 31, w = tid >> 5;
  stage_w_frag(W0, 30, 64, sW0, 32, 64, tid, 256);
  stage_w_frag(W1, 64, 64, sW1, 64, 64, tid, 256);
  stage_w_frag(W2, 64, 30, sW2, 64, 32, tid, 256);
  stage_b(b0, 64, sB0, 64, tid, 256);
  stage_b(b1, 64, sB1, 64, tid, 256);
  stage_b(b2, 30, sB2, 32, tid, 256);
  __syncthreads();
  int g0 = blockIdx.x * 128 + w * 16;
  if (lane < 16) {
    long long g = g0 + lane;
    long long base = (g >> 11) << 11;
    float acc[30];
    for (int d = 0; d < 30; ++d) acc[d] = hIn[g * 30 + d];
    for (int s = 0; s < 8; ++s) {
      long long nb = base + knn[g * 8 + s];
      for (int d = 0; d < 30; ++d) acc[d] += hIn[nb * 30 + d];
    }
    _Float16* row = &tA[w][lane * 32];
    for (int d = 0; d < 30; ++d) row[d] = (_Float16)acc[d];
    row[30] = (_Float16)0.f; row[31] = (_Float16)0.f;
  }
  __syncthreads();
  ffn_layer16(&tA[w][0], 32, sW0, sB0, 64, 64, true, &tB[w][0], nullptr, 0, lane);
  __syncthreads();
  ffn_layer16(&tB[w][0], 64, sW1, sB1, 64, 64, true, &tA[w][0], nullptr, 0, lane);
  __syncthreads();
  ffn_layer16(&tA[w][0], 64, sW2, sB2, 32, 30, false, nullptr,
              hOut + (long long)g0 * 30, 30, lane);
}

// ---------------------------------------------------------------------------
// Kernel 7: global add pool (sum of 2048 rows of 30 per batch)
// ---------------------------------------------------------------------------
__global__ __launch_bounds__(256) void k_sumpool(const float* __restrict__ hIn,
                                                 float* __restrict__ outp) {
  __shared__ float sred[8][30];
  int b = blockIdx.x, tid = threadIdx.x, lane = tid & 31, w = tid >> 5;
  float acc[30];
  for (int d = 0; d < 30; ++d) acc[d] = 0.f;
  for (int n = tid; n < 2048; n += 256) {
    long long g = (long long)b * 2048 + n;
    for (int d = 0; d < 30; ++d) acc[d] += hIn[g * 30 + d];
  }
  for (int d = 0; d < 30; ++d)
    for (int off = 16; off > 0; off >>= 1) acc[d] += __shfl_xor(acc[d], off, 32);
  if (lane == 0)
    for (int d = 0; d < 30; ++d) sred[w][d] = acc[d];
  __syncthreads();
  if (tid < 30) {
    float s = 0.f;
    for (int ww = 0; ww < 8; ++ww) s += sred[ww][tid];
    outp[b * 30 + tid] = s;
  }
}

// ---------------------------------------------------------------------------
// Kernel 8: mpdown FFN: concat(h, aggr[b]) (60) -> 64 -> 64 -> 30, WMMA
// ---------------------------------------------------------------------------
__global__ __launch_bounds__(256) void k_mpdown(
    const float* __restrict__ hIn, const float* __restrict__ aggr,
    const float* W0, const float* b0, const float* W1, const float* b1,
    const float* W2, const float* b2, float* __restrict__ hOut) {
  __shared__ __attribute__((aligned(32))) _Float16 sW0[64 * 64], sW1[64 * 64], sW2[64 * 32];
  __shared__ float sB0[64], sB1[64], sB2[32];
  __shared__ __attribute__((aligned(32))) _Float16 tA[8][16 * 64], tB[8][16 * 64];
  int tid = threadIdx.x, lane = tid & 31, w = tid >> 5;
  stage_w_frag(W0, 60, 64, sW0, 64, 64, tid, 256);
  stage_w_frag(W1, 64, 64, sW1, 64, 64, tid, 256);
  stage_w_frag(W2, 64, 30, sW2, 64, 32, tid, 256);
  stage_b(b0, 64, sB0, 64, tid, 256);
  stage_b(b1, 64, sB1, 64, tid, 256);
  stage_b(b2, 30, sB2, 32, tid, 256);
  __syncthreads();
  int g0 = blockIdx.x * 128 + w * 16;
  if (lane < 16) {
    long long g = g0 + lane;
    int b = (int)(g >> 11);
    _Float16* row = &tA[w][lane * 64];
    for (int d = 0; d < 30; ++d) row[d] = (_Float16)hIn[g * 30 + d];
    for (int d = 0; d < 30; ++d) row[30 + d] = (_Float16)aggr[b * 30 + d];
    row[60] = row[61] = row[62] = row[63] = (_Float16)0.f;
  }
  __syncthreads();
  ffn_layer16(&tA[w][0], 64, sW0, sB0, 64, 64, true, &tB[w][0], nullptr, 0, lane);
  __syncthreads();
  ffn_layer16(&tB[w][0], 64, sW1, sB1, 64, 64, true, &tA[w][0], nullptr, 0, lane);
  __syncthreads();
  ffn_layer16(&tA[w][0], 64, sW2, sB2, 32, 30, false, nullptr,
              hOut + (long long)g0 * 30, 30, lane);
}

// ---------------------------------------------------------------------------
// Kernel 9: final FFN (35 -> 64 -> 64 -> 1, leaky on ALL layers)
// ---------------------------------------------------------------------------
__global__ __launch_bounds__(64) void k_final(
    const float* __restrict__ rgan, const float* __restrict__ cond,
    const float* __restrict__ ggn,
    const float* W0, const float* b0, const float* W1, const float* b1,
    const float* W2, const float* b2, float* __restrict__ out) {
  __shared__ float fin[35], h1[64], h2[64];
  int b = blockIdx.x, t = threadIdx.x;
  if (t < 4) fin[t] = rgan[b * 4 + t];
  if (t == 4) fin[4] = cond[b];
  if (t >= 5 && t < 35) fin[t] = ggn[b * 30 + (t - 5)];
  __syncthreads();
  float v = b0[t];
  for (int i = 0; i < 35; ++i) v += fin[i] * W0[i * 64 + t];
  h1[t] = leaky(v);
  __syncthreads();
  float v2 = b1[t];
  for (int i = 0; i < 64; ++i) v2 += h1[i] * W1[i * 64 + t];
  h2[t] = leaky(v2);
  __syncthreads();
  if (t == 0) {
    float o = b2[0];
    for (int i = 0; i < 64; ++i) o += h2[i] * W2[i];
    out[b] = leaky(o);   // final_linear=False
  }
}

// ---------------------------------------------------------------------------
// Host side
// ---------------------------------------------------------------------------
extern "C" void kernel_launch(void* const* d_in, const int* in_sizes, int n_in,
                              void* d_out, int out_size, void* d_ws, size_t ws_size,
                              hipStream_t stream) {
  (void)n_in; (void)out_size; (void)ws_size;

  // Resolve the flattening order of the nested params dict from in_sizes.
  int iX = 0, iCond = 1;
  int iConv, iRgan, iLin, iGin1, iMp1, iGin2, iMp2, iFin;
  if (in_sizes[0] == 196608) {
    iX = 0; iCond = 1;
    if (in_sizes[9] == 2240) {       // jax-sorted params: conv,final,gin1,gin2,lin,mp1,mp2,rgan
      iConv = 3; iFin = 9; iGin1 = 15; iGin2 = 21; iLin = 27; iMp1 = 33; iMp2 = 39; iRgan = 45;
    } else {                          // insertion order
      iConv = 3; iRgan = 9; iLin = 13; iGin1 = 19; iMp1 = 25; iGin2 = 31; iMp2 = 37; iFin = 43;
    }
  } else {                            // fully sorted leaves: batch,cond,params...,x
    iCond = 1; iConv = 2; iFin = 8; iGin1 = 14; iGin2 = 20; iLin = 26; iMp1 = 32; iMp2 = 38;
    iRgan = 44; iX = 48;
  }
#define P(i) ((const float*)d_in[(i)])
  const float* x    = P(iX);
  const float* cond = P(iCond);

  char* ws = (char*)d_ws;
  float*     feat = (float*)(ws + 0);                       //  8 KB
  float*     rgan = (float*)(ws + 8  * 1024);               //  0.5 KB
  float*     aggr = (float*)(ws + 9  * 1024);               //  3.75 KB
  float*     ggn  = (float*)(ws + 13 * 1024);               //  3.75 KB
  float*     sq   = (float*)(ws + 17 * 1024);               // 256 KB
  int*       idx  = (int*)  (ws + 273 * 1024);              //   2 MB
  _Float16*  h16  = (_Float16*)(ws + 2321 * 1024);          //   4 MB
  float*     hA   = (float*)(ws + 6417 * 1024);             // 7.5 MB
  float*     hB   = (float*)(ws + 14097 * 1024);            // 7.5 MB
  float*     out  = (float*)d_out;                          // [32]

  // 1. conv MLP + maxpool -> feat[32,64]
  k_conv_max<<<32, 256, 0, stream>>>(x, P(iConv), P(iConv+1), P(iConv+2),
                                     P(iConv+3), P(iConv+4), P(iConv+5), feat);
  // 2. rgan -> [32,4]
  k_rgan<<<32, 64, 0, stream>>>(feat, P(iRgan), P(iRgan+1), P(iRgan+2), P(iRgan+3), rgan);
  // 3. lin FFN per node -> hA[65536,30]
  k_lin<<<512, 256, 0, stream>>>(x, cond, rgan, P(iLin), P(iLin+1), P(iLin+2),
                                 P(iLin+3), P(iLin+4), P(iLin+5), hA);
  // 4. f16 copy + squared norms
  k_prep<<<256, 256, 0, stream>>>(hA, h16, sq);
  // 5. kNN (WMMA Gram + top-8) -> idx[65536,8]
  k_knn<<<4096, 256, 0, stream>>>(h16, sq, idx);
  // 6. gin1: hA -> hB
  k_gin<<<512, 256, 0, stream>>>(hA, idx, P(iGin1), P(iGin1+1), P(iGin1+2),
                                 P(iGin1+3), P(iGin1+4), P(iGin1+5), hB);
  // 7. aggr1 = sum(hB)
  k_sumpool<<<32, 256, 0, stream>>>(hB, aggr);
  // 8. mpdown1: (hA, aggr) -> hB
  k_mpdown<<<512, 256, 0, stream>>>(hA, aggr, P(iMp1), P(iMp1+1), P(iMp1+2),
                                    P(iMp1+3), P(iMp1+4), P(iMp1+5), hB);
  // 9. gin2: hB -> hA
  k_gin<<<512, 256, 0, stream>>>(hB, idx, P(iGin2), P(iGin2+1), P(iGin2+2),
                                 P(iGin2+3), P(iGin2+4), P(iGin2+5), hA);
  // 10. aggr2 = sum(hA)
  k_sumpool<<<32, 256, 0, stream>>>(hA, aggr);
  // 11. mpdown2: (hB, aggr2) -> hA
  k_mpdown<<<512, 256, 0, stream>>>(hB, aggr, P(iMp2), P(iMp2+1), P(iMp2+2),
                                    P(iMp2+3), P(iMp2+4), P(iMp2+5), hA);
  // 12. ggn = sum(hA)
  k_sumpool<<<32, 256, 0, stream>>>(hA, ggn);
  // 13. final -> out[32]
  k_final<<<32, 64, 0, stream>>>(rgan, cond, ggn, P(iFin), P(iFin+1), P(iFin+2),
                                 P(iFin+3), P(iFin+4), P(iFin+5), out);
#undef P
}
